// lstmPoem_41918880809329
// MI455X (gfx1250) — matrix operand order repaired and checked by reading, not compiled, not run on hardware
//
#include <hip/hip_runtime.h>
#include <hip/hip_bf16.h>
#include <stdint.h>

#define VOCAB 10000
#define EMB   128
#define HID   128
#define CELL  128
#define SEQ   128
#define BATCH 64
#define KDIM  (EMB + HID)   // 256
#define XT_STRIDE 264       // 256 + 8 bf16 pad -> conflict-free LDS rows

typedef __attribute__((ext_vector_type(16))) __bf16 v16bf;
typedef __attribute__((ext_vector_type(8)))  __bf16 v8bf;
typedef __attribute__((ext_vector_type(8)))  float  v8f;
typedef __attribute__((ext_vector_type(4)))  float  v4f;

union AFrag { v16bf v; v8bf h[2]; };
union Acc8  { v8f v; v4f q[2]; float f[8]; };

__device__ __forceinline__ unsigned short f2bf(float x) {
  unsigned u = __float_as_uint(x);
  unsigned r = (u + 0x7FFFu + ((u >> 16) & 1u)) >> 16;   // RNE
  return (unsigned short)r;
}
__device__ __forceinline__ float sigmoid_f(float x) {
  return 1.0f / (1.0f + __expf(-x));
}
__device__ __forceinline__ float tanh_f(float x) {
  return 1.0f - 2.0f / (__expf(2.0f * x) + 1.0f);
}

// ---------------- phase 1: f32 -> bf16 conversion (output W only) ----------------
__global__ void cvt_f32_bf16(const float* __restrict__ src,
                             unsigned short* __restrict__ dst, int n) {
  int i = blockIdx.x * blockDim.x + threadIdx.x;
  int stride = gridDim.x * blockDim.x;
  for (; i < n; i += stride) dst[i] = f2bf(src[i]);
}

// ---------------- phase 2: sequential LSTM recurrence (1 WG, weights in LDS) ----------------
// Dynamic LDS layout (ushorts):
//   wgF/wgI/wgO/wgC : 4 x [128][264] bf16  (gate weights, padded rows)
//   xT              : [64][264] bf16       (x_h transposed: [b][k])
// total = (4*128*264 + 64*264) * 2 = 304,128 bytes  (< 320 KB WGP LDS)
#define SMEM_USHORTS (4 * CELL * XT_STRIDE + BATCH * XT_STRIDE)

__global__ __launch_bounds__(1024)
void lstm_recurrence(const int* __restrict__ x,
                     const float* __restrict__ emb,
                     const float* __restrict__ Wc,
                     const float* __restrict__ Wf,
                     const float* __restrict__ Wi,
                     const float* __restrict__ Wo,
                     const float* __restrict__ Bc,
                     const float* __restrict__ Bf,
                     const float* __restrict__ Bi,
                     const float* __restrict__ Bo,
                     unsigned short* __restrict__ Hall,   // [SEQ*BATCH][HID] bf16
                     float* __restrict__ outHt,           // [HID][BATCH]
                     float* __restrict__ outCt)           // [CELL][BATCH]
{
  extern __shared__ unsigned short smem[];
  unsigned short* wgF = smem;
  unsigned short* wgI = wgF + CELL * XT_STRIDE;
  unsigned short* wgO = wgI + CELL * XT_STRIDE;
  unsigned short* wgC = wgO + CELL * XT_STRIDE;
  unsigned short* xT  = wgC + CELL * XT_STRIDE;

  const int tid  = threadIdx.x;
  const int lane = tid & 31;
  const int wave = tid >> 5;        // 0..31
  const int lo   = lane & 15;
  const int hi   = lane >> 4;
  const int mt   = wave >> 2;       // 0..7 -> gate rows 16*mt..
  const int nq   = wave & 3;        // 0..3 -> batch cols 16*nq..
  const int mrow = mt * 16;
  const int ncol = nq * 16;

  // ---- one-time: stage gate weights in LDS (f32 -> bf16, padded rows) ----
  for (int i = tid; i < CELL * KDIM; i += 1024) {
    const int r = i >> 8;                 // KDIM == 256
    const int c = i & (KDIM - 1);
    const int o = r * XT_STRIDE + c;
    wgF[o] = f2bf(Wf[i]);
    wgI[o] = f2bf(Wi[i]);
    wgO[o] = f2bf(Wo[i]);
    wgC[o] = f2bf(Wc[i]);
  }
  for (int i = tid; i < BATCH * XT_STRIDE; i += 1024) xT[i] = 0;  // H0 = 0

  // Loop-invariant bias fragments (rows mrow+8*hi .. +7), C/D layout.
  const int bofs = mrow + 8 * hi;
  const v8f biasC = *(const v8f*)(Bc + bofs);
  const v8f biasF = *(const v8f*)(Bf + bofs);
  const v8f biasI = *(const v8f*)(Bi + bofs);
  const v8f biasO = *(const v8f*)(Bo + bofs);

  v8f cacc = v8f{0,0,0,0,0,0,0,0};   // cell state: registers for all 128 steps

  for (int s = 0; s < SEQ; ++s) {
    // ---- embedding gather: xT[b][k<128] = bf16(emb[x[s,b]][k]) ----
    {
      const int b = tid >> 4;
      const int k = (tid & 15) * 8;
      const float* er = emb + (size_t)x[s * BATCH + b] * EMB + k;
      unsigned short* dst = xT + b * XT_STRIDE + k;
      const v4f e0 = *(const v4f*)(er);
      const v4f e1 = *(const v4f*)(er + 4);
      dst[0] = f2bf(e0[0]); dst[1] = f2bf(e0[1]);
      dst[2] = f2bf(e0[2]); dst[3] = f2bf(e0[3]);
      dst[4] = f2bf(e1[0]); dst[5] = f2bf(e1[1]);
      dst[6] = f2bf(e1[2]); dst[7] = f2bf(e1[3]);
    }
    __syncthreads();

    // ---- 4 gate GEMMs; this wave: one 16x16 tile per gate, K = 256 ----
    v8f gF = biasF, gI = biasI, gO = biasO, gC = biasC;

    #pragma unroll
    for (int k0 = 0; k0 < KDIM; k0 += 32) {
      const v16bf bfrag =
          *(const v16bf*)(xT + (ncol + lo) * XT_STRIDE + k0 + 16 * hi);
      const int rb = (mrow + lo) * XT_STRIDE + k0;
      AFrag a;

      a.h[0] = *(const v8bf*)(wgF + rb + 16 * hi);
      a.h[1] = *(const v8bf*)(wgF + rb + 16 * hi + 8);
      gF = __builtin_amdgcn_wmma_f32_16x16x32_bf16(false, a.v, false, bfrag, (short)0, gF, false, false);

      a.h[0] = *(const v8bf*)(wgI + rb + 16 * hi);
      a.h[1] = *(const v8bf*)(wgI + rb + 16 * hi + 8);
      gI = __builtin_amdgcn_wmma_f32_16x16x32_bf16(false, a.v, false, bfrag, (short)0, gI, false, false);

      a.h[0] = *(const v8bf*)(wgO + rb + 16 * hi);
      a.h[1] = *(const v8bf*)(wgO + rb + 16 * hi + 8);
      gO = __builtin_amdgcn_wmma_f32_16x16x32_bf16(false, a.v, false, bfrag, (short)0, gO, false, false);

      a.h[0] = *(const v8bf*)(wgC + rb + 16 * hi);
      a.h[1] = *(const v8bf*)(wgC + rb + 16 * hi + 8);
      gC = __builtin_amdgcn_wmma_f32_16x16x32_bf16(false, a.v, false, bfrag, (short)0, gC, false, false);
    }
    __syncthreads();   // all xT reads done before H region is rewritten

    // ---- elementwise LSTM update; H -> LDS (next step) + global Hall (bf16) ----
    {
      v8f c = cacc;
      Acc8 hv;
      #pragma unroll
      for (int r = 0; r < 8; ++r) {
        const float fg = sigmoid_f(gF[r]);
        const float ig = sigmoid_f(gI[r]);
        const float og = sigmoid_f(gO[r]);
        const float cc = tanh_f(gC[r]);
        const float cn = fg * c[r] + ig * cc;
        c[r] = cn;
        hv.f[r] = tanh_f(cn) * og;
      }
      cacc = c;

      uint4 hp;
      hp.x = (unsigned)f2bf(hv.f[0]) | ((unsigned)f2bf(hv.f[1]) << 16);
      hp.y = (unsigned)f2bf(hv.f[2]) | ((unsigned)f2bf(hv.f[3]) << 16);
      hp.z = (unsigned)f2bf(hv.f[4]) | ((unsigned)f2bf(hv.f[5]) << 16);
      hp.w = (unsigned)f2bf(hv.f[6]) | ((unsigned)f2bf(hv.f[7]) << 16);

      const int bcol  = ncol + lo;            // batch column
      const int hbase = mrow + 8 * hi;        // 8 consecutive h values
      *(uint4*)(xT + bcol * XT_STRIDE + EMB + hbase) = hp;                    // next-step input
      *(uint4*)(Hall + ((size_t)(s * BATCH + bcol)) * HID + hbase) = hp;      // projection input

      if (s == SEQ - 1) {
        #pragma unroll
        for (int r = 0; r < 8; ++r) {
          outHt[(hbase + r) * BATCH + bcol] = hv.f[r];
          outCt[(hbase + r) * BATCH + bcol] = c[r];
        }
      }
    }
  }
}

// ---------------- phase 3: output projection GEMM [10000x128]*[128x8192] ----------------
__global__ __launch_bounds__(256)
void out_proj(const unsigned short* __restrict__ Wbf,   // [VOCAB][HID] bf16
              const unsigned short* __restrict__ Hall,  // [SEQ*BATCH][HID] bf16
              const float* __restrict__ bias,           // [VOCAB]
              float* __restrict__ out)                  // [SEQ*BATCH][VOCAB]
{
  const int lane  = threadIdx.x & 31;
  const int wave  = threadIdx.x >> 5;
  const int lo    = lane & 15;
  const int hi    = lane >> 4;
  const int mtile = blockIdx.x * 8 + wave;
  if (mtile >= VOCAB / 16) return;            // wave-uniform (EXEC stays all-1s)
  const int mrow  = mtile * 16;
  const int ncolB = blockIdx.y * 64;

  __builtin_prefetch(Hall + (size_t)(ncolB + lo) * HID, 0, 1);  // global_prefetch_b8

  AFrag a[4];                                  // W rows, K = 4 chunks of 32
  #pragma unroll
  for (int kc = 0; kc < 4; ++kc) {
    const unsigned short* wp = Wbf + (size_t)(mrow + lo) * HID + kc * 32;
    a[kc].h[0] = *(const v8bf*)(wp + 16 * hi);
    a[kc].h[1] = *(const v8bf*)(wp + 16 * hi + 8);
  }
  const v8f bfr = *(const v8f*)(bias + mrow + 8 * hi);

  #pragma unroll
  for (int nt = 0; nt < 4; ++nt) {
    const int col = ncolB + nt * 16 + lo;      // GEMM column = s*BATCH + b
    const unsigned short* hp = Hall + (size_t)col * HID + 16 * hi;
    v8f acc = bfr;
    #pragma unroll
    for (int kc = 0; kc < 4; ++kc) {
      const v16bf bf = *(const v16bf*)(hp + kc * 32);
      acc = __builtin_amdgcn_wmma_f32_16x16x32_bf16(false, a[kc].v, false, bf, (short)0, acc, false, false);
    }
    Acc8 r; r.v = acc;
    float* op = out + (size_t)col * VOCAB + mrow + 8 * hi;  // 8 consecutive v
    *(v4f*)(op)     = r.q[0];
    *(v4f*)(op + 4) = r.q[1];
  }
}

// ---------------- host side ----------------
extern "C" void kernel_launch(void* const* d_in, const int* in_sizes, int n_in,
                              void* d_out, int out_size, void* d_ws, size_t ws_size,
                              hipStream_t stream) {
  const int*   x   = (const int*)d_in[0];
  const float* emb = (const float*)d_in[1];
  const float* Wc  = (const float*)d_in[2];
  const float* Bc  = (const float*)d_in[3];
  const float* Wf  = (const float*)d_in[4];
  const float* Bf  = (const float*)d_in[5];
  const float* Wi  = (const float*)d_in[6];
  const float* Bi  = (const float*)d_in[7];
  const float* Wo  = (const float*)d_in[8];
  const float* Bo  = (const float*)d_in[9];
  const float* W   = (const float*)d_in[10];
  const float* b   = (const float*)d_in[11];
  float* out = (float*)d_out;

  char* ws = (char*)d_ws;
  unsigned short* wW   = (unsigned short*)(ws);                          // 10000*128 bf16
  unsigned short* hall = (unsigned short*)(ws + (size_t)VOCAB * HID * 2);// 8192*128 bf16

  cvt_f32_bf16<<<1024, 256, 0, stream>>>(W, wW, VOCAB * HID);

  float* outHt = out + (size_t)SEQ * BATCH * VOCAB;
  float* outCt = outHt + (size_t)HID * BATCH;
  const size_t smem_bytes = (size_t)SMEM_USHORTS * sizeof(unsigned short); // 304,128 B
  lstm_recurrence<<<1, 1024, smem_bytes, stream>>>(x, emb, Wc, Wf, Wi, Wo,
                                                   Bc, Bf, Bi, Bo, hall, outHt, outCt);

  out_proj<<<dim3((VOCAB / 16 + 7) / 8, (SEQ * BATCH) / 64), 256, 0, stream>>>(wW, hall, b, out);
}
